// Transformer_68341519614158
// MI455X (gfx1250) — matrix operand order repaired and checked by reading, not compiled, not run on hardware
//
#include <hip/hip_runtime.h>
#include <stdint.h>

// Problem constants (match reference)
#define BB      4
#define SSEQ    1024
#define DD      512
#define HH      8
#define NLAYER  2
#define DFFN    2048
#define MTOT    (BB * SSEQ)    // 4096 token rows
#define HD      (HH * DD)      // 4096 concat width
#define LN_EPS  1e-5f

typedef __attribute__((ext_vector_type(16))) __bf16 v16bf;
typedef __attribute__((ext_vector_type(8)))  float  v8f;

union FragU { uint32_t u[8]; v16bf v; };

__device__ __forceinline__ uint16_t f2b(float f) {
    uint32_t u = __float_as_uint(f);
    return (uint16_t)((u + 0x7FFFu + ((u >> 16) & 1u)) >> 16);   // RNE bf16
}

// ---------------------------------------------------------------------------
// Fragment loaders per CDNA5 ISA VGPR layouts (CDNA5 ISA)
// A: 16x32 bf16, row-major, lanes 0-15 rows M=0..15 (lanes 16-31 same rows,
//    upper K half). VGPR j holds K pair (j<4 ? 2j : 2j+8) + 8*half.
// B: 32x16 bf16 taken from W[N][K] row-major (B = W^T). Lane -> column N,
//    VGPR j holds K pair (j<4 ? 2j : 2j+8) + 8*half (same as A) -> pair loads from W rows.
// ---------------------------------------------------------------------------
__device__ __forceinline__ v16bf load_frag_a(const uint16_t* __restrict__ A, int lda) {
    const int lane = threadIdx.x & 31;
    const int hlf  = lane >> 4;
    const uint16_t* row = A + (size_t)(lane & 15) * lda;
    FragU f;
#pragma unroll
    for (int j = 0; j < 8; ++j) {
        const int kk = ((j < 4) ? (2 * j) : (2 * j + 8)) + 8 * hlf;
        f.u[j] = *reinterpret_cast<const uint32_t*>(row + kk);
    }
    return f.v;
}

__device__ __forceinline__ v16bf load_frag_b(const uint16_t* __restrict__ W, int ldw) {
    const int lane = threadIdx.x & 31;
    const int hlf  = lane >> 4;
    const uint16_t* row = W + (size_t)(lane & 15) * ldw;
    FragU f;
#pragma unroll
    for (int j = 0; j < 8; ++j) {
        const int kk = ((j < 4) ? (2 * j) : (2 * j + 8)) + 8 * hlf;
        f.u[j] = *reinterpret_cast<const uint32_t*>(row + kk);
    }
    return f.v;
}

// ---------------------------------------------------------------------------
// Batched WMMA GEMM: C = act(A[M,K]_bf16 * W[N,K]_bf16^T + bias[N])
// Block = 8 waves, tile 64(M) x 128(N); wave tile 16 x 64 (4 accumulators).
// TSTORE stores C transposed as C[n*ldc + m] (bf16) -> used to build V^T.
// ---------------------------------------------------------------------------
template<bool OUT_BF16, bool RELU, bool TSTORE>
__global__ __launch_bounds__(256) void gemm_wmma(
    const uint16_t* __restrict__ A, int lda, long long strideA,
    const uint16_t* __restrict__ W, int ldw, long long strideW,
    const float* __restrict__ bias, long long strideBias,
    void* __restrict__ Cout, long long strideC, int ldc, int K)
{
    const int bz = blockIdx.z;
    const uint16_t* Ab = A + (size_t)bz * strideA;
    const uint16_t* Wb = W + (size_t)bz * strideW;
    const float* biasb = bias ? (bias + (size_t)bz * strideBias) : nullptr;

    const int wid  = threadIdx.x >> 5;
    const int lane = threadIdx.x & 31;
    const int m0 = blockIdx.y * 64  + (wid & 3) * 16;
    const int n0 = blockIdx.x * 128 + (wid >> 2) * 64;

    v8f acc[4] = {};
    for (int k0 = 0; k0 < K; k0 += 32) {
        v16bf a = load_frag_a(Ab + (size_t)m0 * lda + k0, lda);
#pragma unroll
        for (int nt = 0; nt < 4; ++nt) {
            v16bf b = load_frag_b(Wb + (size_t)(n0 + nt * 16) * ldw + k0, ldw);
            acc[nt] = __builtin_amdgcn_wmma_f32_16x16x32_bf16(
                false, a, false, b, (short)0, acc[nt], false, false);
        }
    }

    const int hlf = lane >> 4;
    const int nl  = lane & 15;
#pragma unroll
    for (int nt = 0; nt < 4; ++nt) {
        const int n = n0 + nt * 16 + nl;
        const float bv = biasb ? biasb[n] : 0.0f;
#pragma unroll
        for (int r = 0; r < 8; ++r) {
            const int m = m0 + r + 8 * hlf;
            float v = acc[nt][r] + bv;
            if (RELU) v = fmaxf(v, 0.0f);
            if (TSTORE) {
                uint16_t* C = (uint16_t*)Cout + (size_t)bz * strideC;
                C[(size_t)n * ldc + m] = f2b(v);
            } else if (OUT_BF16) {
                uint16_t* C = (uint16_t*)Cout + (size_t)bz * strideC;
                C[(size_t)m * ldc + n] = f2b(v);
            } else {
                float* C = (float*)Cout + (size_t)bz * strideC;
                C[(size_t)m * ldc + n] = v;
            }
        }
    }
}

// ---------------------------------------------------------------------------
// Fused attention: one block per (b, h, 16-row tile of S).
// Phase 1: scores tile 16 x S in registers (wave w owns columns [w*128,..)).
// Phase 2: softmax (scale 1/sqrt(dk)=0.125) via shfl_xor + LDS cross-wave.
// Phase 3: probs -> LDS bf16 (16 x 1024, 32KB).
// Phase 4: P*V via WMMA, A from LDS, B from V^T in global; write concat.
// ---------------------------------------------------------------------------
__global__ __launch_bounds__(256) void attn_kernel(
    const uint16_t* __restrict__ Qb,   // [H][MTOT][DD]
    const uint16_t* __restrict__ Kb,   // [H][MTOT][DD]
    const uint16_t* __restrict__ Vt,   // [H][DD][MTOT]  (transposed)
    uint16_t* __restrict__ catb)       // [MTOT][HD]
{
    __shared__ uint16_t attnLds[16][SSEQ];   // 32 KB
    __shared__ float redA[8][16];
    __shared__ float redB[8][16];

    const int s0 = blockIdx.x * 16;
    const int h  = blockIdx.y;
    const int b  = blockIdx.z;
    const int wid  = threadIdx.x >> 5;
    const int lane = threadIdx.x & 31;
    const int hlf  = lane >> 4;
    const int nl   = lane & 15;

    const uint16_t* Qh = Qb + (size_t)h * MTOT * DD + (size_t)(b * SSEQ + s0) * DD;
    const uint16_t* Kh = Kb + (size_t)h * MTOT * DD + (size_t)(b * SSEQ) * DD;
    const uint16_t* Vh = Vt + (size_t)h * DD * MTOT;

    // Phase 1: Q * K^T (K matrix is already W-form [t][e])
    v8f acc[8] = {};
    const int nbase = wid * 128;
    for (int k0 = 0; k0 < DD; k0 += 32) {
        v16bf a = load_frag_a(Qh + k0, DD);
#pragma unroll
        for (int f = 0; f < 8; ++f) {
            v16bf bf = load_frag_b(Kh + (size_t)(nbase + f * 16) * DD + k0, DD);
            acc[f] = __builtin_amdgcn_wmma_f32_16x16x32_bf16(
                false, a, false, bf, (short)0, acc[f], false, false);
        }
    }

    // Phase 2: softmax.  scale by 1/sqrt(dk), dk = D/H = 64.
    const float scale = 0.125f;
#pragma unroll
    for (int f = 0; f < 8; ++f)
#pragma unroll
        for (int r = 0; r < 8; ++r) acc[f][r] *= scale;

    float rmax[8];
#pragma unroll
    for (int r = 0; r < 8; ++r) {
        float m = acc[0][r];
#pragma unroll
        for (int f = 1; f < 8; ++f) m = fmaxf(m, acc[f][r]);
        m = fmaxf(m, __shfl_xor(m, 1));
        m = fmaxf(m, __shfl_xor(m, 2));
        m = fmaxf(m, __shfl_xor(m, 4));
        m = fmaxf(m, __shfl_xor(m, 8));
        rmax[r] = m;       // row (r + 8*hlf) partial max for this wave
    }
    if (lane == 0)  { for (int r = 0; r < 8; ++r) redA[wid][r]     = rmax[r]; }
    if (lane == 16) { for (int r = 0; r < 8; ++r) redA[wid][8 + r] = rmax[r]; }
    __syncthreads();
    float gmax[8];
#pragma unroll
    for (int r = 0; r < 8; ++r) {
        const int row = r + 8 * hlf;
        float m = redA[0][row];
        for (int w = 1; w < 8; ++w) m = fmaxf(m, redA[w][row]);
        gmax[r] = m;
    }
    float rsum[8];
#pragma unroll
    for (int r = 0; r < 8; ++r) {
        float s = 0.0f;
#pragma unroll
        for (int f = 0; f < 8; ++f) {
            const float e = __expf(acc[f][r] - gmax[r]);
            acc[f][r] = e;
            s += e;
        }
        s += __shfl_xor(s, 1);
        s += __shfl_xor(s, 2);
        s += __shfl_xor(s, 4);
        s += __shfl_xor(s, 8);
        rsum[r] = s;
    }
    if (lane == 0)  { for (int r = 0; r < 8; ++r) redB[wid][r]     = rsum[r]; }
    if (lane == 16) { for (int r = 0; r < 8; ++r) redB[wid][8 + r] = rsum[r]; }
    __syncthreads();
#pragma unroll
    for (int r = 0; r < 8; ++r) {
        const int row = r + 8 * hlf;
        float s = 0.0f;
        for (int w = 0; w < 8; ++w) s += redB[w][row];
        const float inv = 1.0f / s;
#pragma unroll
        for (int f = 0; f < 8; ++f) acc[f][r] *= inv;
    }

    // Phase 3: probabilities -> LDS (row-major 16 x 1024 bf16)
#pragma unroll
    for (int f = 0; f < 8; ++f)
#pragma unroll
        for (int r = 0; r < 8; ++r)
            attnLds[r + 8 * hlf][nbase + f * 16 + nl] = f2b(acc[f][r]);
    __syncthreads();

    // Phase 4: heads = P * V, wave w owns output columns [w*64, w*64+64)
    v8f hacc[4] = {};
    const int ebase = wid * 64;
    const uint16_t* arow = &attnLds[nl][0];
    for (int k0 = 0; k0 < SSEQ; k0 += 32) {
        FragU fa;
#pragma unroll
        for (int j = 0; j < 8; ++j) {
            const int kk = ((j < 4) ? (2 * j) : (2 * j + 8)) + 8 * hlf;
            fa.u[j] = *reinterpret_cast<const uint32_t*>(arow + k0 + kk);
        }
#pragma unroll
        for (int f = 0; f < 4; ++f) {
            v16bf bf = load_frag_b(
                Vh + (size_t)(ebase + f * 16) * MTOT + (size_t)b * SSEQ + k0, MTOT);
            hacc[f] = __builtin_amdgcn_wmma_f32_16x16x32_bf16(
                false, fa.v, false, bf, (short)0, hacc[f], false, false);
        }
    }
#pragma unroll
    for (int f = 0; f < 4; ++f) {
        const int e = ebase + f * 16 + nl;
#pragma unroll
        for (int r = 0; r < 8; ++r) {
            const int s = s0 + r + 8 * hlf;
            catb[(size_t)(b * SSEQ + s) * HD + h * DD + e] = f2b(hacc[f][r]);
        }
    }
}

// ---------------------------------------------------------------------------
// Residual + LayerNorm over rows of D=512; block = 256 threads = 1 row.
// Writes f32 x (and optionally bf16 x for the next GEMM).
// ---------------------------------------------------------------------------
__global__ __launch_bounds__(256) void ln_residual(
    const float* __restrict__ xin, const float* __restrict__ delta,
    const float* __restrict__ g, const float* __restrict__ beta,
    float* __restrict__ dstF, uint16_t* __restrict__ dstB)
{
    __shared__ float sred[256];
    const int row = blockIdx.x;
    const int t = threadIdx.x;
    const float* xr = xin   + (size_t)row * DD;
    const float* dr = delta + (size_t)row * DD;
    const float v0 = xr[t]       + dr[t];
    const float v1 = xr[t + 256] + dr[t + 256];

    sred[t] = v0 + v1;
    __syncthreads();
    for (int o = 128; o > 0; o >>= 1) {
        if (t < o) sred[t] += sred[t + o];
        __syncthreads();
    }
    const float mean = sred[0] * (1.0f / DD);
    __syncthreads();
    const float d0 = v0 - mean, d1 = v1 - mean;
    sred[t] = d0 * d0 + d1 * d1;
    __syncthreads();
    for (int o = 128; o > 0; o >>= 1) {
        if (t < o) sred[t] += sred[t + o];
        __syncthreads();
    }
    const float inv = rsqrtf(sred[0] * (1.0f / DD) + LN_EPS);

    const float y0 = d0 * inv * g[t]       + beta[t];
    const float y1 = d1 * inv * g[t + 256] + beta[t + 256];
    dstF[(size_t)row * DD + t]       = y0;
    dstF[(size_t)row * DD + t + 256] = y1;
    if (dstB) {
        dstB[(size_t)row * DD + t]       = f2b(y0);
        dstB[(size_t)row * DD + t + 256] = f2b(y1);
    }
}

// ---------------------------------------------------------------------------
// Elementwise helpers
// ---------------------------------------------------------------------------
__global__ void cvt_f32_bf16(const float* __restrict__ src,
                             uint16_t* __restrict__ dst, long long n) {
    long long i = (long long)blockIdx.x * blockDim.x + threadIdx.x;
    const long long stride = (long long)gridDim.x * blockDim.x;
    for (; i < n; i += stride) dst[i] = f2b(src[i]);
}
__global__ void copy_f32(const float* __restrict__ src,
                         float* __restrict__ dst, long long n) {
    long long i = (long long)blockIdx.x * blockDim.x + threadIdx.x;
    const long long stride = (long long)gridDim.x * blockDim.x;
    for (; i < n; i += stride) dst[i] = src[i];
}

// ---------------------------------------------------------------------------
extern "C" void kernel_launch(void* const* d_in, const int* in_sizes, int n_in,
                              void* d_out, int out_size, void* d_ws, size_t ws_size,
                              hipStream_t stream) {
    (void)in_sizes; (void)n_in; (void)out_size; (void)ws_size;
    const float* x   = (const float*)d_in[0];
    const float* Wq  = (const float*)d_in[1];
    const float* bq  = (const float*)d_in[2];
    const float* Wk  = (const float*)d_in[3];
    const float* bk  = (const float*)d_in[4];
    const float* Wv  = (const float*)d_in[5];
    const float* bv  = (const float*)d_in[6];
    const float* Wo  = (const float*)d_in[7];
    const float* bo  = (const float*)d_in[8];
    const float* W1  = (const float*)d_in[9];
    const float* b1  = (const float*)d_in[10];
    const float* W2  = (const float*)d_in[11];
    const float* b2  = (const float*)d_in[12];
    const float* ln1g = (const float*)d_in[13];
    const float* ln1b = (const float*)d_in[14];
    const float* ln2g = (const float*)d_in[15];
    const float* ln2b = (const float*)d_in[16];

    // bump allocator on workspace (~215 MB total)
    char* ws = (char*)d_ws;
    size_t off = 0;
    auto alloc = [&](size_t bytes) -> char* {
        off = (off + 255) & ~(size_t)255;
        char* p = ws + off;
        off += bytes;
        return p;
    };
    float*    xf   = (float*)   alloc((size_t)MTOT * DD * 4);
    float*    tmp  = (float*)   alloc((size_t)MTOT * DD * 4);
    uint16_t* xb   = (uint16_t*)alloc((size_t)MTOT * DD * 2);
    uint16_t* qb   = (uint16_t*)alloc((size_t)HH * MTOT * DD * 2);
    uint16_t* kb   = (uint16_t*)alloc((size_t)HH * MTOT * DD * 2);
    uint16_t* vt   = (uint16_t*)alloc((size_t)HH * DD * MTOT * 2);
    uint16_t* catb = (uint16_t*)alloc((size_t)MTOT * HD * 2);
    uint16_t* hb   = (uint16_t*)alloc((size_t)MTOT * DFFN * 2);
    uint16_t* wqb  = (uint16_t*)alloc((size_t)NLAYER * HH * DD * DD * 2);
    uint16_t* wkb  = (uint16_t*)alloc((size_t)NLAYER * HH * DD * DD * 2);
    uint16_t* wvb  = (uint16_t*)alloc((size_t)NLAYER * HH * DD * DD * 2);
    uint16_t* wob  = (uint16_t*)alloc((size_t)NLAYER * DD * HD * 2);
    uint16_t* w1b  = (uint16_t*)alloc((size_t)NLAYER * DFFN * DD * 2);
    uint16_t* w2b  = (uint16_t*)alloc((size_t)NLAYER * DD * DFFN * 2);

    const dim3 blk(256);
    const dim3 cvtg(2048);
    const long long nX  = (long long)MTOT * DD;
    const long long nWq = (long long)NLAYER * HH * DD * DD;
    copy_f32   <<<cvtg, blk, 0, stream>>>(x, xf, nX);
    cvt_f32_bf16<<<cvtg, blk, 0, stream>>>(x,  xb,  nX);
    cvt_f32_bf16<<<cvtg, blk, 0, stream>>>(Wq, wqb, nWq);
    cvt_f32_bf16<<<cvtg, blk, 0, stream>>>(Wk, wkb, nWq);
    cvt_f32_bf16<<<cvtg, blk, 0, stream>>>(Wv, wvb, nWq);
    cvt_f32_bf16<<<cvtg, blk, 0, stream>>>(Wo, wob, (long long)NLAYER * DD * HD);
    cvt_f32_bf16<<<cvtg, blk, 0, stream>>>(W1, w1b, (long long)NLAYER * DFFN * DD);
    cvt_f32_bf16<<<cvtg, blk, 0, stream>>>(W2, w2b, (long long)NLAYER * DD * DFFN);

    for (int i = 0; i < NLAYER; ++i) {
        const size_t wOff = (size_t)i * HH * DD * DD;
        // Q, K: [H][MTOT][DD] bf16;  V stored transposed: [H][DD][MTOT]
        gemm_wmma<true, false, false><<<dim3(DD / 128, MTOT / 64, HH), blk, 0, stream>>>(
            xb, DD, 0, wqb + wOff, DD, (long long)DD * DD,
            bq + (size_t)i * HH * DD, DD, qb, (long long)MTOT * DD, DD, DD);
        gemm_wmma<true, false, false><<<dim3(DD / 128, MTOT / 64, HH), blk, 0, stream>>>(
            xb, DD, 0, wkb + wOff, DD, (long long)DD * DD,
            bk + (size_t)i * HH * DD, DD, kb, (long long)MTOT * DD, DD, DD);
        gemm_wmma<true, false, true><<<dim3(DD / 128, MTOT / 64, HH), blk, 0, stream>>>(
            xb, DD, 0, wvb + wOff, DD, (long long)DD * DD,
            bv + (size_t)i * HH * DD, DD, vt, (long long)DD * MTOT, MTOT, DD);

        attn_kernel<<<dim3(SSEQ / 16, HH, BB), blk, 0, stream>>>(qb, kb, vt, catb);

        // Output projection: M=4096, N=512, K=4096 -> f32 tmp
        gemm_wmma<false, false, false><<<dim3(DD / 128, MTOT / 64, 1), blk, 0, stream>>>(
            catb, HD, 0, wob + (size_t)i * DD * HD, HD, 0,
            bo + (size_t)i * DD, 0, tmp, 0, DD, HD);
        ln_residual<<<MTOT, blk, 0, stream>>>(
            xf, tmp, ln1g + (size_t)i * DD, ln1b + (size_t)i * DD, xf, xb);

        // FFN
        gemm_wmma<true, true, false><<<dim3(DFFN / 128, MTOT / 64, 1), blk, 0, stream>>>(
            xb, DD, 0, w1b + (size_t)i * DFFN * DD, DD, 0,
            b1 + (size_t)i * DFFN, 0, hb, 0, DFFN, DD);
        gemm_wmma<false, false, false><<<dim3(DD / 128, MTOT / 64, 1), blk, 0, stream>>>(
            hb, DFFN, 0, w2b + (size_t)i * DD * DFFN, DFFN, 0,
            b2 + (size_t)i * DD, 0, tmp, 0, DD, DFFN);

        const bool last = (i == NLAYER - 1);
        ln_residual<<<MTOT, blk, 0, stream>>>(
            xf, tmp, ln2g + (size_t)i * DD, ln2b + (size_t)i * DD,
            last ? (float*)d_out : xf, last ? nullptr : xb);
    }
}
